// DGM_22857815949798
// MI455X (gfx1250) — matrix operand order repaired and checked by reading, not compiled, not run on hardware
//
#include <hip/hip_runtime.h>
#include <hip/hip_bf16.h>

// DGM pipeline for MI455X (gfx1250, wave32, WMMA + TDM).
//
//   K1 (WMMA bf16): h0 = x @ W
//   K2 (atomics):   agg[dst] += h0[src] over edges
//   K3:             h = relu(h0+agg) -> d_out, bf16 copy, row sqnorms
//   K4 (TDM + WMMA):dist[i][j] = sqrt(max(sq_i + sq_j - 2 h_i.h_j, 0))
//   K5 (TDM):       per-row LN stats + entmax-1.5 tau via bisection (row in LDS)
//   K6:             res[i][j] = (i==j) ? diag_i : dist_ij < max(c_i,c_j); logprobs

#define NN 8192
#define DD 256

typedef __attribute__((ext_vector_type(16))) __bf16        v16bf;
typedef __attribute__((ext_vector_type(8)))  float         v8f;
typedef __attribute__((ext_vector_type(4)))  unsigned int  v4u;
typedef __attribute__((ext_vector_type(8)))  int           v8i;
typedef __attribute__((ext_vector_type(4)))  int           v4i;

union FragU { v16bf v; uint4 q[2]; };

// ---- Tensor Data Mover: 2D tile (rows x dim0 elements) global -> LDS ----
// dsz: 0=1B,1=2B,2=4B elements. Issued once per calling wave; EXEC-independent.
__device__ __forceinline__ void tdm_load_2d(unsigned lds_addr, const void* gptr,
                                            unsigned dsz, unsigned dim0,
                                            unsigned rows, unsigned stride0,
                                            unsigned trows) {
  const unsigned long long ga = (unsigned long long)gptr;
  v4u g0;
  g0[0] = 1u;                                   // count=1, is_restore=0
  g0[1] = lds_addr;                             // lds_addr [63:32]
  g0[2] = (unsigned)ga;                         // global_addr lo
  g0[3] = (unsigned)((ga >> 32) & 0x01FFFFFFull) | (2u << 30);  // addr hi | type=2
  v8i g1;
  g1[0] = (int)(dsz << 16);                     // wg_mask=0 | data_size
  g1[1] = (int)((dim0 & 0xFFFFu) << 16);        // tensor_dim0 lo16 at [63:48]
  g1[2] = (int)((dim0 >> 16) | ((trows & 0xFFFFu) << 16));  // dim0 hi | tensor_dim1 lo
  g1[3] = (int)((trows >> 16) | (dim0 << 16));  // tensor_dim1 hi | tile_dim0
  g1[4] = (int)rows;                            // tile_dim1 | tile_dim2=0
  g1[5] = (int)stride0;                         // tensor_dim0_stride lo32
  g1[6] = 0;                                    // stride0 hi16 | stride1 lo16
  g1[7] = 0;
  const v4i z = {0, 0, 0, 0};
#if __has_include(<hip/amd_detail/amd_gfx1250_TDM.h>)
  const v8i z8 = {0, 0, 0, 0, 0, 0, 0, 0};
  __builtin_amdgcn_tensor_load_to_lds(g0, g1, z, z, z8, 0);
#else
  __builtin_amdgcn_tensor_load_to_lds(g0, g1, z, z, 0);
#endif
}

// ---------------- K1: h0 = x @ W (WMMA bf16, f32 accum) ----------------
__global__ __launch_bounds__(128) void k_gemm_xw(const float* __restrict__ x,
                                                 const float* __restrict__ W,
                                                 float* __restrict__ h0) {
  const int lane = threadIdx.x & 31;
  const int wave = threadIdx.x >> 5;
  const int half = lane >> 4;
  const int mr   = lane & 15;
  const int ibase = blockIdx.x * 16;
  const int jt    = blockIdx.y * 64 + wave * 16;
  v8f acc = {};
  const float4* xr = (const float4*)(x + (ibase + mr) * DD);
#pragma unroll
  for (int kq = 0; kq < DD / 32; ++kq) {
    const int kb = kq * 32;
    // A fragment: two contiguous 8-float runs per lane (ISA 7.12.2 bf16 A layout)
    const float4 f0 = xr[kq * 8 + 2 * half];
    const float4 f1 = xr[kq * 8 + 2 * half + 1];
    const float4 f2 = xr[kq * 8 + 4 + 2 * half];
    const float4 f3 = xr[kq * 8 + 4 + 2 * half + 1];
    v16bf a, b;
    a[0] = (__bf16)f0.x; a[1] = (__bf16)f0.y; a[2]  = (__bf16)f0.z; a[3]  = (__bf16)f0.w;
    a[4] = (__bf16)f1.x; a[5] = (__bf16)f1.y; a[6]  = (__bf16)f1.z; a[7]  = (__bf16)f1.w;
    a[8] = (__bf16)f2.x; a[9] = (__bf16)f2.y; a[10] = (__bf16)f2.z; a[11] = (__bf16)f2.w;
    a[12] = (__bf16)f3.x; a[13] = (__bf16)f3.y; a[14] = (__bf16)f3.z; a[15] = (__bf16)f3.w;
#pragma unroll
    for (int i = 0; i < 16; ++i)   // B: column gather (W row-major), K = kb+(i&7)+8*half+16*(i>>3)
      b[i] = (__bf16)W[(kb + (i & 7) + 8 * half + 16 * (i >> 3)) * DD + (jt + mr)];
    acc = __builtin_amdgcn_wmma_f32_16x16x32_bf16(false, a, false, b, (short)0, acc,
                                                  false, false);
  }
#pragma unroll
  for (int r = 0; r < 8; ++r)
    h0[(ibase + r + 8 * half) * DD + (jt + mr)] = acc[r];
}

// ---------------- K2: edge scatter-add (global_atomic_add_f32) ----------------
__global__ void k_scatter(const float* __restrict__ h0, const int* __restrict__ edges,
                          int nE, float* __restrict__ agg) {
  const long long t = (long long)blockIdx.x * blockDim.x + threadIdx.x;
  const int per = DD / 4;
  const long long total = (long long)nE * per;
  if (t >= total) return;
  const int e = (int)(t / per);
  const int f = (int)(t % per) * 4;
  const int s = edges[e];
  const int d = edges[nE + e];
  const float4 v = *(const float4*)(h0 + (long long)s * DD + f);
  float* a = agg + (long long)d * DD + f;
  atomicAdd(a + 0, v.x);
  atomicAdd(a + 1, v.y);
  atomicAdd(a + 2, v.z);
  atomicAdd(a + 3, v.w);
}

// ---------------- K3: relu combine + bf16 copy + row sqnorm ----------------
__global__ __launch_bounds__(256) void k_relu(const float* __restrict__ h0,
                                              const float* __restrict__ agg,
                                              float* __restrict__ hout,
                                              __bf16* __restrict__ hbf,
                                              float* __restrict__ sq) {
  __shared__ float red[256];
  const int i = blockIdx.x;
  const int f = threadIdx.x;
  float v = h0[i * DD + f] + agg[i * DD + f];
  v = fmaxf(v, 0.0f);
  hout[i * DD + f] = v;
  hbf[i * DD + f] = (__bf16)v;
  red[f] = v * v;
  __syncthreads();
  for (int s2 = 128; s2 > 0; s2 >>= 1) {
    if (f < s2) red[f] += red[f + s2];
    __syncthreads();
  }
  if (f == 0) sq[i] = red[0];
}

// ---------------- K4: pairwise distances via TDM-staged WMMA ----------------
// block 256 thr = 8 waves -> 32(i) x 64(j) macro tile.
__global__ __launch_bounds__(256) void k_dist(const __bf16* __restrict__ hbf,
                                              const float* __restrict__ sq,
                                              float* __restrict__ dist) {
  __shared__ __align__(16) __bf16 As[32 * DD];  // 16 KB
  __shared__ __align__(16) __bf16 Bs[64 * DD];  // 32 KB
  const int tid  = threadIdx.x;
  const int lane = tid & 31, wave = tid >> 5;
  const int half = lane >> 4, mr = lane & 15;
  const int isub = wave >> 2, jsub = wave & 3;
  const int ibase = blockIdx.y * 32;
  const int jbase = blockIdx.x * 64;

  if (wave == 0) {  // wave 0 drives both tile DMAs; TENSORcnt tracks completion
    tdm_load_2d((unsigned)(unsigned long long)(void*)As,
                hbf + (long long)ibase * DD, 1, DD, 32, DD, NN);
    tdm_load_2d((unsigned)(unsigned long long)(void*)Bs,
                hbf + (long long)jbase * DD, 1, DD, 64, DD, NN);
    __builtin_amdgcn_s_wait_tensorcnt(0);
  }
  __syncthreads();

  // Per-lane fragment runs are K-contiguous -> 2x ds_load_b128 per fragment.
  const uint4* arow = (const uint4*)(As + (isub * 16 + mr) * DD);
  const uint4* brow = (const uint4*)(Bs + (jsub * 16 + mr) * DD);
  v8f acc = {};
#pragma unroll
  for (int kq = 0; kq < DD / 32; ++kq) {
    FragU a, b;
    a.q[0] = arow[kq * 4 + half];          // K = 32kq + 8*half .. +7
    a.q[1] = arow[kq * 4 + 2 + half];      // K = 32kq + 16 + 8*half .. +7
    b.q[0] = brow[kq * 4 + half];          // K = 32kq + 8*half .. +7
    b.q[1] = brow[kq * 4 + 2 + half];      // K = 32kq + 16 + 8*half .. +7
    acc = __builtin_amdgcn_wmma_f32_16x16x32_bf16(false, a.v, false, b.v, (short)0,
                                                  acc, false, false);
  }
#pragma unroll
  for (int r = 0; r < 8; ++r) {
    const int gi = ibase + isub * 16 + r + 8 * half;
    const int gj = jbase + jsub * 16 + mr;
    const float d2 = sq[gi] + sq[gj] - 2.0f * acc[r];
    dist[(long long)gi * NN + gj] = sqrtf(fmaxf(d2, 0.0f));
  }
}

// ---------------- K5: per-row LN stats + entmax tau (bisection) ----------------
__global__ __launch_bounds__(256) void k_rowstats(const float* __restrict__ dist,
                                                  const float* __restrict__ gamma_p,
                                                  const float* __restrict__ beta_p,
                                                  float* __restrict__ c,
                                                  int* __restrict__ dflag) {
  __shared__ __align__(16) float row[NN];   // 32 KB, TDM-filled
  __shared__ float red[256];
  const int i = blockIdx.x;
  const int tid = threadIdx.x;
  const float gamma = gamma_p[0], beta = beta_p[0];

  if (tid < 32) {  // wave 0: DMA the whole row into LDS
    tdm_load_2d((unsigned)(unsigned long long)(void*)row,
                dist + (long long)i * NN, 2, NN, 1, NN, NN);
    __builtin_amdgcn_s_wait_tensorcnt(0);
  }
  __syncthreads();

  float sum = 0.f, sumsq = 0.f;
#pragma unroll
  for (int q = 0; q < 8; ++q) {
    const float4 v = ((const float4*)row)[tid + q * 256];
    sum   += v.x + v.y + v.z + v.w;
    sumsq += v.x * v.x + v.y * v.y + v.z * v.z + v.w * v.w;
  }
  red[tid] = sum; __syncthreads();
  for (int st = 128; st > 0; st >>= 1) { if (tid < st) red[tid] += red[tid + st]; __syncthreads(); }
  const float sumAll = red[0]; __syncthreads();
  red[tid] = sumsq; __syncthreads();
  for (int st = 128; st > 0; st >>= 1) { if (tid < st) red[tid] += red[tid + st]; __syncthreads(); }
  const float sumsqAll = red[0]; __syncthreads();

  // probs = -dist; mean_p = -mean_d; var negation-invariant; ddof=1
  const float meand = sumAll / (float)NN;
  const float var   = (sumsqAll - (float)NN * meand * meand) / (float)(NN - 1);
  const float stdv  = sqrtf(fmaxf(var, 0.f));
  const float scale = gamma / (stdv + 1e-6f);
  // zn_ij = scale*(meand - d_ij) + beta (j!=i); zn_ii = -1e-6; zhalf = zn/2.

  float mx = -1e30f;
  for (int j = tid; j < NN; j += 256) {
    const float zh = (j == i) ? -0.5e-6f : 0.5f * (scale * (meand - row[j]) + beta);
    mx = fmaxf(mx, zh);
  }
  red[tid] = mx; __syncthreads();
  for (int st = 128; st > 0; st >>= 1) { if (tid < st) red[tid] = fmaxf(red[tid], red[tid + st]); __syncthreads(); }
  const float m = red[0]; __syncthreads();

  // g(t) = sum_j [zhalf_j - m - t]_+^2 strictly decreasing; tau* in [-1, 0].
  float lo = -1.0f, hi = 0.0f;
  for (int it = 0; it < 30; ++it) {
    const float t = 0.5f * (lo + hi);
    float gs = 0.f;
    for (int j = tid; j < NN; j += 256) {
      const float zh = (j == i) ? -0.5e-6f : 0.5f * (scale * (meand - row[j]) + beta);
      const float u = fmaxf(zh - m - t, 0.f);
      gs += u * u;
    }
    red[tid] = gs; __syncthreads();
    for (int st = 128; st > 0; st >>= 1) { if (tid < st) red[tid] += red[tid + st]; __syncthreads(); }
    const float gtot = red[0]; __syncthreads();
    if (gtot > 1.0f) lo = t; else hi = t;
  }
  if (tid == 0) {
    const float tau   = 0.5f * (lo + hi);
    const float theta = 2.0f * (tau + m);  // p_ij > 0  <=>  zn_ij > theta
    c[i]     = meand - (theta - beta) * (stdv + 1e-6f) / gamma;  // dist cutoff
    dflag[i] = (-1e-6f > theta) ? 1 : 0;
  }
}

// ---------------- K6: symmetrized adjacency + logprobs ----------------
__global__ __launch_bounds__(256) void k_adj(const float* __restrict__ dist,
                                             const float* __restrict__ c,
                                             const int* __restrict__ dflag,
                                             int* __restrict__ res,
                                             int* __restrict__ logprobs) {
  __shared__ int red[256];
  const int i = blockIdx.x;
  const int tid = threadIdx.x;
  const float ci = c[i];
  int cnt = 0;
  for (int j = tid; j < NN; j += 256) {
    int v;
    if (j == i) v = dflag[i];
    else        v = (dist[(long long)i * NN + j] < fmaxf(ci, c[j])) ? 1 : 0;
    res[(long long)i * NN + j] = v;
    cnt += v;
  }
  red[tid] = cnt; __syncthreads();
  for (int st = 128; st > 0; st >>= 1) { if (tid < st) red[tid] += red[tid + st]; __syncthreads(); }
  if (tid == 0) logprobs[i] = red[0];
}

extern "C" void kernel_launch(void* const* d_in, const int* in_sizes, int n_in,
                              void* d_out, int out_size, void* d_ws, size_t ws_size,
                              hipStream_t stream) {
  const float* x     = (const float*)d_in[0];
  const float* W     = (const float*)d_in[1];
  const float* gamma = (const float*)d_in[2];
  const float* beta  = (const float*)d_in[3];
  const int*   edges = (const int*)d_in[4];
  const int    nE    = in_sizes[4] / 2;

  // outputs: h [NN*DD] f32, res [NN*NN] i32, logprobs [NN] i32, concatenated
  float* out      = (float*)d_out;
  float* h_out    = out;
  int*   res      = (int*)(out + (size_t)NN * DD);
  int*   logprobs = res + (size_t)NN * NN;

  // workspace layout (float units)
  float*  ws   = (float*)d_ws;
  float*  h0   = ws;                            // NN*DD
  float*  agg  = h0 + (size_t)NN * DD;          // NN*DD
  float*  p    = agg + (size_t)NN * DD;
  __bf16* hbf  = (__bf16*)p;                    // NN*DD bf16 == NN*DD/2 floats
  float*  sq   = p + (size_t)NN * DD / 2;       // NN
  float*  cthr = sq + NN;                       // NN
  int*    dfl  = (int*)(cthr + NN);             // NN
  float*  dist = (float*)(dfl + NN);            // NN*NN

  hipMemsetAsync(agg, 0, (size_t)NN * DD * sizeof(float), stream);

  k_gemm_xw<<<dim3(NN / 16, DD / 64), 128, 0, stream>>>(x, W, h0);

  {
    const long long total = (long long)nE * (DD / 4);
    const int blocks = (int)((total + 255) / 256);
    k_scatter<<<blocks, 256, 0, stream>>>(h0, edges, nE, agg);
  }

  k_relu<<<NN, 256, 0, stream>>>(h0, agg, h_out, hbf, sq);

  k_dist<<<dim3(NN / 64, NN / 32), 256, 0, stream>>>(hbf, sq, dist);

  k_rowstats<<<NN, 256, 0, stream>>>(dist, gamma, beta, cthr, dfl);

  k_adj<<<NN, 256, 0, stream>>>(dist, cthr, dfl, res, logprobs);
}
